// ContrastiveLossV2_65335042507212
// MI455X (gfx1250) — matrix operand order repaired and checked by reading, not compiled, not run on hardware
//
#include <hip/hip_runtime.h>
#include <math.h>

typedef _Float16 half4  __attribute__((ext_vector_type(4)));
typedef _Float16 half8  __attribute__((ext_vector_type(8)));
typedef _Float16 half16 __attribute__((ext_vector_type(16)));
typedef float    f32x8  __attribute__((ext_vector_type(8)));

#define NROWS 8192      // 2N
#define DDIM  128
#define NTILES_N 4                      // 16x64 strip per wave
#define STRIPS_M (NROWS / 16)           // 512
#define STRIPS_N (NROWS / (16 * NTILES_N)) // 128
#define NSTRIP   (STRIPS_M * STRIPS_N)  // 65536 waves
#define GEMM_WAVES_PER_BLOCK 4
#define GEMM_BLOCKS (NSTRIP / GEMM_WAVES_PER_BLOCK) // 16384

// ---------------------------------------------------------------------------
// Kernel 1: row-wise L2 normalize (x / max(||x||, 1e-12)).
// One wave per row of 128 floats; emit f16 reps (row-major, A operand) and
// f16 repsT (transposed copy; the WMMA kernel reads its B operand from reps).
// Both f16 copies (4 MB total) stay resident in the 192 MB L2.
// ---------------------------------------------------------------------------
__global__ __launch_bounds__(256)
void cl_normalize_kernel(const float* __restrict__ emb_i,
                         const float* __restrict__ emb_j,
                         _Float16* __restrict__ reps,
                         _Float16* __restrict__ repsT) {
    const int wave = threadIdx.x >> 5;
    const int lane = threadIdx.x & 31;
    const int row  = blockIdx.x * 8 + wave;
    if (row >= NROWS) return;

    const float* src = (row < (NROWS / 2))
                         ? (emb_i + (size_t)row * DDIM)
                         : (emb_j + (size_t)(row - NROWS / 2) * DDIM);

    const float4 v = *(const float4*)(src + lane * 4);
    float ss = v.x * v.x + v.y * v.y + v.z * v.z + v.w * v.w;
    #pragma unroll
    for (int m = 16; m >= 1; m >>= 1) ss += __shfl_xor(ss, m, 32);

    const float scale = 1.0f / fmaxf(sqrtf(ss), 1e-12f);

    half4 h;
    h.x = (_Float16)(v.x * scale);
    h.y = (_Float16)(v.y * scale);
    h.z = (_Float16)(v.z * scale);
    h.w = (_Float16)(v.w * scale);

    *(half4*)(reps + (size_t)row * DDIM + lane * 4) = h;

    const int k0 = lane * 4;
    repsT[(size_t)(k0 + 0) * NROWS + row] = h.x;
    repsT[(size_t)(k0 + 1) * NROWS + row] = h.y;
    repsT[(size_t)(k0 + 2) * NROWS + row] = h.z;
    repsT[(size_t)(k0 + 3) * NROWS + row] = h.w;
}

// ---------------------------------------------------------------------------
// Kernel 2: one wave per 16x64 strip of sim = R @ R^T.
// 4 accumulator tiles, K=128 via 4 K-steps x 4 tiles = 16x
// v_wmma_f32_16x16x32_f16 per wave; A loaded once per K-step, reused 4x.
// class_pairs (268 MB, read once) is prefetched then loaded NON-TEMPORAL so
// the HBM stream does not evict the L2-resident f16 operands.
// Fused (sim - class_pairs)^2 with a single cross-lane reduction per wave.
//
// A operand (16x32 f16, MxK) per-lane layout (ISA 7.12.2):
//   lanes 0-15 : row M=lane,    elems 0..7 -> K=kk+0..7,  8..15 -> K=kk+16..23
//   lanes 16-31: row M=lane-16, elems 0..7 -> K=kk+8..15, 8..15 -> K=kk+24..31
// B operand (32x16 f16, KxN): lane holds column N=lane&15, elems -> K in the same order as A.
// C/D (16x16 f32): VGPR r, lanes 0-15 -> (M=r, N=lane&15), lanes 16-31 -> (M=8+r).
// ---------------------------------------------------------------------------
__global__ __launch_bounds__(GEMM_WAVES_PER_BLOCK * 32)
void cl_simstrip_kernel(const _Float16* __restrict__ reps,
                        const _Float16* __restrict__ repsT,
                        const float* __restrict__ cls,
                        float* __restrict__ partials) {
    __shared__ float wsum[GEMM_WAVES_PER_BLOCK];

    const int lane = threadIdx.x & 31;
    const int winb = threadIdx.x >> 5;
    const int wid  = blockIdx.x * GEMM_WAVES_PER_BLOCK + winb;
    const int sm   = wid / STRIPS_N;          // strip row   (0..511)
    const int sn   = wid % STRIPS_N;          // strip col   (0..127)
    const int row0 = sm << 4;                 // 16 rows
    const int col0 = sn << 6;                 // 64 cols

    const int hsel = lane >> 4;               // wave half
    const int lrow = lane & 15;

    const _Float16* Ar = reps  + (size_t)(row0 + lrow) * DDIM + (hsel << 3);
    const _Float16* Br = reps  + (size_t)(col0 + lrow) * DDIM + (hsel << 3);

    // --- prefetch the 16x64 f32 class_pairs tile (the HBM stream) ---------
    // rows: hsel splits 0-7 / 8-15; cols: 16 lanes x 16B stride = 256B/row.
    {
        const float* pf = cls + (size_t)(row0 + hsel * 8) * NROWS + col0 + lrow * 4;
        #pragma unroll
        for (int r = 0; r < 8; ++r)
            __builtin_prefetch(pf + (size_t)r * NROWS, 0, 0);
    }

    // --- WMMA strip: A reused across 4 N-tiles per K-step ------------------
    f32x8 c[NTILES_N] = {};
    #pragma unroll
    for (int kk = 0; kk < DDIM; kk += 32) {
        half8 a0 = *(const half8*)(Ar + kk);        // K = kk + 8*hsel + 0..7
        half8 a1 = *(const half8*)(Ar + kk + 16);   // K = kk + 16 + 8*hsel + 0..7
        half16 a = __builtin_shufflevector(a0, a1,
                     0, 1, 2, 3, 4, 5, 6, 7, 8, 9, 10, 11, 12, 13, 14, 15);
        const _Float16* Bk = Br + kk;
        #pragma unroll
        for (int t = 0; t < NTILES_N; ++t) {
            half8 b0 = *(const half8*)(Bk + (size_t)t * 16 * DDIM);
            half8 b1 = *(const half8*)(Bk + (size_t)t * 16 * DDIM + 16);
            half16 b = __builtin_shufflevector(b0, b1,
                         0, 1, 2, 3, 4, 5, 6, 7, 8, 9, 10, 11, 12, 13, 14, 15);
            c[t] = __builtin_amdgcn_wmma_f32_16x16x32_f16(
                       false, a, false, b, (short)0, c[t], false, false);
        }
    }

    // --- fused MSE over the strip (non-temporal stream reads) --------------
    const int   mbase = row0 + (hsel << 3);
    const float* cp0  = cls + (size_t)mbase * NROWS + col0 + lrow;
    float acc = 0.0f;
    #pragma unroll
    for (int t = 0; t < NTILES_N; ++t) {
        #pragma unroll
        for (int r = 0; r < 8; ++r) {
            const float cp = __builtin_nontemporal_load(cp0 + (size_t)r * NROWS + t * 16);
            const float d  = c[t][r] - cp;
            acc = fmaf(d, d, acc);
        }
    }
    #pragma unroll
    for (int m = 16; m >= 1; m >>= 1) acc += __shfl_xor(acc, m, 32);

    if (lane == 0) wsum[winb] = acc;
    __syncthreads();
    if (threadIdx.x == 0) {
        float s = 0.0f;
        #pragma unroll
        for (int w = 0; w < GEMM_WAVES_PER_BLOCK; ++w) s += wsum[w];
        partials[blockIdx.x] = s;
    }
}

// ---------------------------------------------------------------------------
// Kernel 3: deterministic final reduction over block partials, mean + store.
// ---------------------------------------------------------------------------
__global__ __launch_bounds__(256)
void cl_reduce_kernel(const float* __restrict__ partials, float* __restrict__ out) {
    __shared__ float sh[256];
    float s = 0.0f;
    for (int i = threadIdx.x; i < GEMM_BLOCKS; i += 256) s += partials[i];
    sh[threadIdx.x] = s;
    __syncthreads();
    #pragma unroll
    for (int step = 128; step >= 1; step >>= 1) {
        if ((int)threadIdx.x < step) sh[threadIdx.x] += sh[threadIdx.x + step];
        __syncthreads();
    }
    if (threadIdx.x == 0)
        out[0] = sh[0] * (1.0f / ((float)NROWS * (float)NROWS));
}

// ---------------------------------------------------------------------------
extern "C" void kernel_launch(void* const* d_in, const int* in_sizes, int n_in,
                              void* d_out, int out_size, void* d_ws, size_t ws_size,
                              hipStream_t stream) {
    (void)in_sizes; (void)n_in; (void)out_size; (void)ws_size;

    const float* emb_i = (const float*)d_in[0];   // [4096,128] f32
    const float* emb_j = (const float*)d_in[1];   // [4096,128] f32
    const float* cls   = (const float*)d_in[2];   // [8192,8192] f32

    char* ws = (char*)d_ws;
    _Float16* reps     = (_Float16*)(ws);                           // 2 MB
    _Float16* repsT    = (_Float16*)(ws + (size_t)2 * 1024 * 1024); // 2 MB
    float*    partials = (float*)   (ws + (size_t)4 * 1024 * 1024); // 64 KB

    // 1) normalize: 8192 rows, 8 waves (8 rows) per 256-thread block
    cl_normalize_kernel<<<NROWS / 8, 256, 0, stream>>>(emb_i, emb_j, reps, repsT);

    // 2) WMMA strips + fused MSE partials: 65536 strips, 4 waves/block
    cl_simstrip_kernel<<<GEMM_BLOCKS, GEMM_WAVES_PER_BLOCK * 32, 0, stream>>>(
        reps, repsT, cls, partials);

    // 3) deterministic scalar reduce
    cl_reduce_kernel<<<1, 256, 0, stream>>>(partials, (float*)d_out);
}
